// MultiheadMaskedCrossAttention_30648886624909
// MI455X (gfx1250) — compile-verified, matrix operand order checked by reading, not run on hardware
//
#include <hip/hip_runtime.h>
#include <hip/hip_bf16.h>

// ---------------------------------------------------------------------------
// MultiheadMaskedCrossAttention for MI455X (gfx1250, wave32, WMMA + TDM)
// B=2, Nq=2048, Nk=4096, C=512, H=8, Dh=64
// ---------------------------------------------------------------------------

typedef __attribute__((ext_vector_type(8)))  _Float16 v8h;
typedef __attribute__((ext_vector_type(16))) _Float16 v16h;
typedef __attribute__((ext_vector_type(8)))  float    v8f;
typedef __attribute__((ext_vector_type(4)))  unsigned int u32x4;
typedef __attribute__((ext_vector_type(8)))  int      i32x8;
typedef __attribute__((ext_vector_type(4)))  int      i32x4;

#define DEV __device__ __forceinline__

static constexpr int B_  = 2;
static constexpr int NQ  = 2048;
static constexpr int NK  = 4096;
static constexpr int C_  = 512;
static constexpr int H_  = 8;
static constexpr int DH  = 64;
static constexpr float SCALE = 0.125f;   // 1/sqrt(64)
static constexpr float NEGF  = -1e30f;

#if __has_builtin(__builtin_amdgcn_tensor_load_to_lds) && \
    __has_builtin(__builtin_amdgcn_s_wait_tensorcnt)
#define HAVE_TDM 1
#else
#define HAVE_TDM 0
#endif

// ---- WMMA operand loader --------------------------------------------------
// CDNA5 16-bit A-layout (CDNA5 ISA): lane holds row (m = lane&15);
// lanes 0-15: K halves {0..7, 16..23}; lanes 16-31: K halves {8..15, 24..31}.
// B operand uses the same per-lane pattern (row ptr = n-th column's K data).
DEV v16h load16ab(const _Float16* p) {
  const int lane = threadIdx.x & 31;
  const _Float16* q = p + ((lane & 16) >> 1);
  v8h lo = *(const v8h*)(q);
  v8h hi = *(const v8h*)(q + 16);
  return __builtin_shufflevector(lo, hi, 0,1,2,3,4,5,6,7,8,9,10,11,12,13,14,15);
}

DEV v8f wmma_f32_f16(v16h a, v16h b, v8f c) {
  return __builtin_amdgcn_wmma_f32_16x16x32_f16(false, a, false, b, (short)0, c,
                                                false, false);
}

DEV float redmax16(float v) {
  v = fmaxf(v, __shfl_xor(v, 1, 32));
  v = fmaxf(v, __shfl_xor(v, 2, 32));
  v = fmaxf(v, __shfl_xor(v, 4, 32));
  v = fmaxf(v, __shfl_xor(v, 8, 32));
  return v;
}
DEV float redsum16(float v) {
  v += __shfl_xor(v, 1, 32);
  v += __shfl_xor(v, 2, 32);
  v += __shfl_xor(v, 4, 32);
  v += __shfl_xor(v, 8, 32);
  return v;
}

#if HAVE_TDM
// ---- Tensor Data Mover: 2-D tile load, global -> LDS (f16 elements) -------
// D# per CDNA5 ISA: group0 {count=1, lds_addr,
// global_addr, type=2}; group1 {data_size=1(2B), tensor_dim0/1, tile_dim0/1,
// tensor_dim0_stride}. Groups 2/3 zero (2-D tensor).
DEV void tdm_load_2d_f16(unsigned lds_off, const void* gaddr,
                         unsigned tile_w, unsigned tile_h,
                         unsigned tensor_w, unsigned tensor_h,
                         unsigned row_stride) {
  unsigned long long ga = (unsigned long long)(uintptr_t)gaddr;
  u32x4 g0 = {};
  g0[0] = 1u;                                          // count=1, user desc
  g0[1] = lds_off;                                     // lds_addr
  g0[2] = (unsigned)(ga & 0xffffffffu);                // global_addr[31:0]
  g0[3] = (unsigned)((ga >> 32) & 0x1ffffffu) | (2u << 30);  // addr[56:32]|type=2
  i32x8 g1 = {};
  g1[0] = (int)(1u << 16);                             // data_size=1 (2 bytes)
  g1[1] = (int)((tensor_w & 0xffffu) << 16);           // tensor_dim0[15:0]
  g1[2] = (int)(((tensor_w >> 16) & 0xffffu) | ((tensor_h & 0xffffu) << 16));
  g1[3] = (int)(((tensor_h >> 16) & 0xffffu) | ((tile_w & 0xffffu) << 16));
  g1[4] = (int)(tile_h & 0xffffu);                     // tile_dim1 (tile_dim2=0)
  g1[5] = (int)row_stride;                             // tensor_dim0_stride lo
  g1[6] = 0;
  g1[7] = 0;
  i32x4 z4 = {};
#if __clang_major__ >= 23
  i32x8 z8 = {};
  __builtin_amdgcn_tensor_load_to_lds(g0, g1, z4, z4, z8, 0);
#else
  __builtin_amdgcn_tensor_load_to_lds(g0, g1, z4, z4, 0);
#endif
}
#endif

// ---------------------------------------------------------------------------
// Kernel 1: f32 -> f16 conversion (grid-stride)
// ---------------------------------------------------------------------------
__global__ void cvt_f32_f16(const float* __restrict__ src,
                            _Float16* __restrict__ dst, int n) {
  for (int i = blockIdx.x * blockDim.x + threadIdx.x; i < n;
       i += gridDim.x * blockDim.x)
    dst[i] = (_Float16)src[i];
}

// ---------------------------------------------------------------------------
// Kernel 2: Y = X @ W^T + bias. One wave per 16x64 strip (4 acc tiles share
// one A operand: 1.25 loads per WMMA). K = 512, step 32.
// mode 0: Q -> f16 (b,h,q,d) | 1: K -> f16 (b,h,k,d) | 2: V -> f16 (b,h,d,k)
// mode 3: O -> f32 row-major (final output)
// ---------------------------------------------------------------------------
__global__ void gemm_wmma(const _Float16* __restrict__ X,
                          const _Float16* __restrict__ W,
                          const float* __restrict__ bias,
                          void* __restrict__ out, int rows, int mode) {
  const int wave = threadIdx.x >> 5;
  const int lane = threadIdx.x & 31;
  const int gw   = blockIdx.x * (blockDim.x >> 5) + wave;
  const int strips = C_ / 64;                   // 8
  const int rowt = gw / strips;
  const int col0 = (gw % strips) * 64;
  const int row0 = rowt * 16;
  if (row0 >= rows) return;

  const int ln = lane & 15;
  const int hg = lane >> 4;

  const _Float16* Xr = X + (size_t)(row0 + ln) * C_;
  const _Float16* Wr0 = W + (size_t)(col0 + ln) * C_;

  v8f acc[4] = {};
#pragma unroll
  for (int kk = 0; kk < C_; kk += 32) {
    const v16h a = load16ab(Xr + kk);
#pragma unroll
    for (int t = 0; t < 4; ++t) {
      const v16h b = load16ab(Wr0 + (size_t)(t * 16) * C_ + kk);
      acc[t] = wmma_f32_f16(a, b, acc[t]);
    }
  }

#pragma unroll
  for (int t = 0; t < 4; ++t) {
    const int j = col0 + t * 16 + ln;
    const float bj = bias[j];
#pragma unroll
    for (int r = 0; r < 8; ++r) {
      const int i = row0 + r + 8 * hg;
      const float v = acc[t][r] + bj;
      if (mode == 3) {
        ((float*)out)[(size_t)i * C_ + j] = v;
      } else {
        const int h = j >> 6, d = j & 63;
        _Float16* o = (_Float16*)out;
        if (mode == 0) {
          const int b = i >> 11, q = i & (NQ - 1);
          o[(((size_t)(b * H_ + h) * NQ) + q) * DH + d] = (_Float16)v;
        } else if (mode == 1) {
          const int b = i >> 12, k = i & (NK - 1);
          o[(((size_t)(b * H_ + h) * NK) + k) * DH + d] = (_Float16)v;
        } else {
          const int b = i >> 12, k = i & (NK - 1);
          o[(((size_t)(b * H_ + h) * DH) + d) * NK + k] = (_Float16)v;
        }
      }
    }
  }
}

// ---------------------------------------------------------------------------
// Kernel 3: flash attention. Block = one (b,h) and 128 q rows (8 waves x 16).
// K/V tiles (32 keys) staged in LDS via TDM (double-buffered), shared by all
// 8 waves -> 8x less global K/V traffic.
// ---------------------------------------------------------------------------
__global__ void attn_wmma(const _Float16* __restrict__ Q,
                          const _Float16* __restrict__ K,
                          const _Float16* __restrict__ Vt,
                          const float* __restrict__ qcoord,
                          const float* __restrict__ kcoord,
                          const int* __restrict__ radius,
                          _Float16* __restrict__ out) {
  __shared__ __align__(16) _Float16 ldsK[2][32 * 64];   // (key, d)   4KB x2
  __shared__ __align__(16) _Float16 ldsV[2][64 * 32];   // (d, key)   4KB x2
  __shared__ __align__(16) _Float16 plds[8][16 * 32];   // P tiles    8KB

  const int wave = threadIdx.x >> 5;
  const int lane = threadIdx.x & 31;
  const int bh     = blockIdx.x >> 4;           // 0..15
  const int qgroup = blockIdx.x & 15;
  const int h = bh & (H_ - 1), b = bh >> 3;
  const int qbase = qgroup * 128 + wave * 16;
  const int ln = lane & 15;
  const int hg = lane >> 4;

  const int rv = *radius;
  const float r2 = (float)rv * (float)rv;

  float qx[8], qy[8], qz[8], qn[8];
#pragma unroll
  for (int r = 0; r < 8; ++r) {
    const int qi = qbase + r + 8 * hg;
    const float* p = qcoord + ((size_t)b * NQ + qi) * 3;
    qx[r] = p[0]; qy[r] = p[1]; qz[r] = p[2];
    qn[r] = qx[r] * qx[r] + qy[r] * qy[r] + qz[r] * qz[r];
  }

  const _Float16* Qb = Q  + ((size_t)bh * NQ + qbase) * DH;
  const _Float16* Kb = K  + (size_t)bh * NK * DH;
  const _Float16* Vb = Vt + (size_t)bh * DH * NK;

  const v16h aq0 = load16ab(Qb + (size_t)ln * DH + 0);
  const v16h aq1 = load16ab(Qb + (size_t)ln * DH + 32);

  v8f acc[4] = {};
  float mrow[8], lrow[8];
#pragma unroll
  for (int r = 0; r < 8; ++r) { mrow[r] = -INFINITY; lrow[r] = 0.0f; }

  _Float16* pl = &plds[wave][0];
  const int ntiles = NK / 32;                   // 128

#if HAVE_TDM
  // prologue: stage tile 0 into buffer 0 (TDM; one issue per block)
  if (wave == 0) {
    tdm_load_2d_f16((unsigned)(uintptr_t)&ldsK[0][0], Kb, 64, 32, 64, 32, 64);
    tdm_load_2d_f16((unsigned)(uintptr_t)&ldsV[0][0], Vb, 32, 64, 32, 64, NK);
  }
#endif

  for (int it = 0; it < ntiles; ++it) {
    const int kb = it * 32;
    const int cur = it & 1;
    __syncthreads();                            // prev consumers done
#if HAVE_TDM
    if (wave == 0) {
      if (it + 1 < ntiles) {                    // prefetch next tile via TDM
        const int nb = (it + 1) & 1;
        const int kn = kb + 32;
        tdm_load_2d_f16((unsigned)(uintptr_t)&ldsK[nb][0],
                        Kb + (size_t)kn * DH, 64, 32, 64, 32, 64);
        tdm_load_2d_f16((unsigned)(uintptr_t)&ldsV[nb][0],
                        Vb + kn, 32, 64, 32, 64, NK);
        __builtin_amdgcn_s_wait_tensorcnt((short)2);   // tile `it` complete
      } else {
        __builtin_amdgcn_s_wait_tensorcnt((short)0);
      }
    }
#else
    {
      const int tid = threadIdx.x;
      // K tile: 32 rows x 64 halves, contiguous in global
      ((v8h*)&ldsK[cur][0])[tid] =
          ((const v8h*)(Kb + (size_t)kb * DH))[tid];
      // V tile: 64 rows x 32 halves, row stride NK
      const int row = tid >> 2, ch = tid & 3;
      *(v8h*)&ldsV[cur][row * 32 + ch * 8] =
          *(const v8h*)(Vb + (size_t)row * NK + kb + ch * 8);
    }
#endif
    __syncthreads();                            // tile `it` visible

    // ---- scores: two 16x16 tiles over 32 keys (K from LDS) ---------------
    v8f s0 = {}, s1 = {};
    {
      const _Float16* kr0 = &ldsK[cur][0] + (size_t)ln * DH;
      const _Float16* kr1 = &ldsK[cur][0] + (size_t)(16 + ln) * DH;
      s0 = wmma_f32_f16(aq0, load16ab(kr0 + 0),  s0);
      s0 = wmma_f32_f16(aq1, load16ab(kr0 + 32), s0);
      s1 = wmma_f32_f16(aq0, load16ab(kr1 + 0),  s1);
      s1 = wmma_f32_f16(aq1, load16ab(kr1 + 32), s1);
    }
    // ---- distance mask: qn + kn - 2 q.k > r^2 means "attend" -------------
    float kx[2], ky[2], kz[2], kn2[2];
#pragma unroll
    for (int t = 0; t < 2; ++t) {
      const int ki = kb + t * 16 + ln;
      const float* p = kcoord + ((size_t)b * NK + ki) * 3;
      kx[t] = p[0]; ky[t] = p[1]; kz[t] = p[2];
      kn2[t] = kx[t] * kx[t] + ky[t] * ky[t] + kz[t] * kz[t];
    }
    float rmax[8];
#pragma unroll
    for (int r = 0; r < 8; ++r) {
      float d20 = qn[r] + kn2[0] -
                  2.0f * (qx[r] * kx[0] + qy[r] * ky[0] + qz[r] * kz[0]);
      float d21 = qn[r] + kn2[1] -
                  2.0f * (qx[r] * kx[1] + qy[r] * ky[1] + qz[r] * kz[1]);
      float v0 = (d20 > r2) ? s0[r] * SCALE : NEGF;
      float v1 = (d21 > r2) ? s1[r] * SCALE : NEGF;
      s0[r] = v0; s1[r] = v1;
      rmax[r] = redmax16(fmaxf(v0, v1));
    }
    // ---- online softmax --------------------------------------------------
#pragma unroll
    for (int r = 0; r < 8; ++r) {
      const float mnew  = fmaxf(mrow[r], rmax[r]);
      const float alpha = __expf(mrow[r] - mnew);
      mrow[r] = mnew;
      const float p0 = __expf(s0[r] - mnew);
      const float p1 = __expf(s1[r] - mnew);
      s0[r] = p0; s1[r] = p1;
      lrow[r] = lrow[r] * alpha + redsum16(p0 + p1);
#pragma unroll
      for (int t = 0; t < 4; ++t) acc[t][r] *= alpha;
    }
    // ---- P (C/D layout) -> f16 A-layout via per-wave LDS round trip ------
#pragma unroll
    for (int r = 0; r < 8; ++r) {
      const int row = r + 8 * hg;
      pl[row * 32 + ln]      = (_Float16)s0[r];
      pl[row * 32 + 16 + ln] = (_Float16)s1[r];
    }
    const v16h ap = load16ab(pl + (size_t)ln * 32);
    // ---- PV: out(16q x 64d) += P(16x32) @ V(32k x 64d), V from LDS -------
#pragma unroll
    for (int t = 0; t < 4; ++t) {
      const _Float16* vr = &ldsV[cur][0] + (size_t)(t * 16 + ln) * 32;
      acc[t] = wmma_f32_f16(ap, load16ab(vr), acc[t]);
    }
  }

  // ---- epilogue: normalize, write (b, q, C) f16 for O-projection ---------
#pragma unroll
  for (int r = 0; r < 8; ++r) {
    const float inv = 1.0f / lrow[r];
    const int q = qbase + r + 8 * hg;
#pragma unroll
    for (int t = 0; t < 4; ++t) {
      const int j = h * DH + t * 16 + ln;
      out[((size_t)b * NQ + q) * C_ + j] = (_Float16)(acc[t][r] * inv);
    }
  }
}

// ---------------------------------------------------------------------------
// Host launcher
// ---------------------------------------------------------------------------
extern "C" void kernel_launch(void* const* d_in, const int* in_sizes, int n_in,
                              void* d_out, int out_size, void* d_ws, size_t ws_size,
                              hipStream_t stream) {
  const float* query_tokens = (const float*)d_in[0];
  const float* key_tokens   = (const float*)d_in[1];
  const float* query_coords = (const float*)d_in[2];
  const float* key_coords   = (const float*)d_in[3];
  const float* Wq = (const float*)d_in[4];
  const float* bq = (const float*)d_in[5];
  const float* Wk = (const float*)d_in[6];
  const float* bk = (const float*)d_in[7];
  const float* Wv = (const float*)d_in[8];
  const float* bv = (const float*)d_in[9];
  const float* Wo = (const float*)d_in[10];
  const float* bo = (const float*)d_in[11];
  const int* radius = (const int*)d_in[12];

  char* ws = (char*)d_ws;
  size_t off = 0;
  auto alloc = [&](size_t bytes) {
    char* p = ws + off;
    off = (off + bytes + 255) & ~(size_t)255;
    return p;
  };
  _Float16* Qf   = (_Float16*)alloc((size_t)B_ * H_ * NQ * DH * 2);
  _Float16* Kf   = (_Float16*)alloc((size_t)B_ * H_ * NK * DH * 2);
  _Float16* Vt   = (_Float16*)alloc((size_t)B_ * H_ * DH * NK * 2);
  _Float16* Xq16 = (_Float16*)alloc((size_t)B_ * NQ * C_ * 2);
  _Float16* Xk16 = (_Float16*)alloc((size_t)B_ * NK * C_ * 2);
  _Float16* Wq16 = (_Float16*)alloc((size_t)C_ * C_ * 2);
  _Float16* Wk16 = (_Float16*)alloc((size_t)C_ * C_ * 2);
  _Float16* Wv16 = (_Float16*)alloc((size_t)C_ * C_ * 2);
  _Float16* Wo16 = (_Float16*)alloc((size_t)C_ * C_ * 2);
  _Float16* AX   = (_Float16*)alloc((size_t)B_ * NQ * C_ * 2);
  (void)ws_size;

  const int nXq = B_ * NQ * C_;
  const int nXk = B_ * NK * C_;
  const int nW  = C_ * C_;
  dim3 blk(256);
  cvt_f32_f16<<<dim3((nXq + 255) / 256), blk, 0, stream>>>(query_tokens, Xq16, nXq);
  cvt_f32_f16<<<dim3((nXk + 255) / 256), blk, 0, stream>>>(key_tokens,   Xk16, nXk);
  cvt_f32_f16<<<dim3((nW  + 255) / 256), blk, 0, stream>>>(Wq, Wq16, nW);
  cvt_f32_f16<<<dim3((nW  + 255) / 256), blk, 0, stream>>>(Wk, Wk16, nW);
  cvt_f32_f16<<<dim3((nW  + 255) / 256), blk, 0, stream>>>(Wv, Wv16, nW);
  cvt_f32_f16<<<dim3((nW  + 255) / 256), blk, 0, stream>>>(Wo, Wo16, nW);

  // projections: one wave per 16x64 strip, 8 waves per block
  const int rowsQ = B_ * NQ;                        // 4096
  const int rowsK = B_ * NK;                        // 8192
  const int wavesQ = (rowsQ / 16) * (C_ / 64);      // 2048
  const int wavesK = (rowsK / 16) * (C_ / 64);      // 4096
  gemm_wmma<<<dim3(wavesQ / 8), blk, 0, stream>>>(Xq16, Wq16, bq, (void*)Qf, rowsQ, 0);
  gemm_wmma<<<dim3(wavesK / 8), blk, 0, stream>>>(Xk16, Wk16, bk, (void*)Kf, rowsK, 1);
  gemm_wmma<<<dim3(wavesK / 8), blk, 0, stream>>>(Xk16, Wv16, bv, (void*)Vt, rowsK, 2);

  // attention: block = (b,h, 128-row q group); 2*8*16 = 256 blocks
  attn_wmma<<<dim3(256), blk, 0, stream>>>(Qf, Kf, Vt, query_coords,
                                           key_coords, radius, AX);

  // output projection -> f32 d_out
  gemm_wmma<<<dim3(wavesQ / 8), blk, 0, stream>>>(AX, Wo16, bo, d_out, rowsQ, 3);
  (void)in_sizes; (void)n_in; (void)out_size;
}
